// PairScorer_3702261809686
// MI455X (gfx1250) — matrix operand order repaired and checked by reading, not compiled, not run on hardware
//
#include <hip/hip_runtime.h>

typedef __attribute__((ext_vector_type(16))) __bf16        v16bf;
typedef __attribute__((ext_vector_type(8)))  float         v8f;
typedef __attribute__((ext_vector_type(8)))  unsigned int  v8u;

#define BATCH       8
#define NSEQ        192
#define DDIM        768
#define HDIM        150
#define HPAD        160
#define K1DIM       2304      // 3*DDIM
#define PAIRS_PER_B 18336     // 192*191/2
#define NTILES      10        // HPAD/16
#define MSUB        4         // 4 x 16-pair sub-tiles per wave (M=64)
#define TOTAL_M64   2292      // 8*18336/64
#define PAD_VAL     (-1000.0f)

// ---------- bf16 split helpers ----------
__device__ __forceinline__ unsigned short bf16_rne(float x) {
  unsigned u = __builtin_bit_cast(unsigned, x);
  unsigned r = u + 0x7FFFu + ((u >> 16) & 1u);
  return (unsigned short)(r >> 16);
}
__device__ __forceinline__ float bf16_to_f32(unsigned short h) {
  unsigned u = ((unsigned)h) << 16;
  return __builtin_bit_cast(float, u);
}

// Pack 16 f32 (A-operand order: f[2d],f[2d+1] -> dword d, per ISA 16-bit
// A-layout) into hi/lo bf16 vectors for the split-bf16 3x scheme.
__device__ __forceinline__ void split_pack16(const float* f, v16bf& ah, v16bf& al) {
  v8u hi, lo;
#pragma unroll
  for (int d = 0; d < 8; ++d) {
    float x0 = f[2 * d], x1 = f[2 * d + 1];
    unsigned short h0 = bf16_rne(x0), h1 = bf16_rne(x1);
    unsigned short l0 = bf16_rne(x0 - bf16_to_f32(h0));
    unsigned short l1 = bf16_rne(x1 - bf16_to_f32(h1));
    hi[d] = (unsigned)h0 | ((unsigned)h1 << 16);
    lo[d] = (unsigned)l0 | ((unsigned)l1 << 16);
  }
  ah = __builtin_bit_cast(v16bf, hi);
  al = __builtin_bit_cast(v16bf, lo);
}

__device__ __forceinline__ void load8(const float* p, float* f) {
  float4 a = *(const float4*)(p);
  float4 b = *(const float4*)(p + 4);
  f[0] = a.x; f[1] = a.y; f[2] = a.z; f[3] = a.w;
  f[4] = b.x; f[5] = b.y; f[6] = b.z; f[7] = b.w;
}

#define WMMA_BF16(A, B, C) \
  __builtin_amdgcn_wmma_f32_16x16x32_bf16(false, (A), false, (B), (short)0, (C), false, false)

// c += a_hi*b_hi + a_hi*b_lo + a_lo*b_hi   (split-bf16, f32 accumulate)
__device__ __forceinline__ v8f wmma3v(v8f c, v16bf ah, v16bf al, v16bf bh, v16bf bl) {
  c = WMMA_BF16(ah, bh, c);
  c = WMMA_BF16(ah, bl, c);
  c = WMMA_BF16(al, bh, c);
  return c;
}

// ---------- weight prep: transpose + pad + bf16 hi/lo split ----------
// W1 (2304x150 row-major) -> w1t[n][k], n in [0,160), k in [0,2304)
__global__ void prep_w1(const float* __restrict__ W1,
                        unsigned short* __restrict__ hi,
                        unsigned short* __restrict__ lo) {
  int idx = blockIdx.x * 256 + threadIdx.x;
  if (idx >= HPAD * K1DIM) return;
  int n = idx / K1DIM, k = idx - n * K1DIM;
  float v = (n < HDIM) ? W1[(size_t)k * HDIM + n] : 0.0f;
  unsigned short h = bf16_rne(v);
  hi[idx] = h;
  lo[idx] = bf16_rne(v - bf16_to_f32(h));
}

// W2 (150x150 row-major) -> w2t[n][k], n,k in [0,160)
__global__ void prep_w2(const float* __restrict__ W2,
                        unsigned short* __restrict__ hi,
                        unsigned short* __restrict__ lo) {
  int idx = blockIdx.x * 256 + threadIdx.x;
  if (idx >= HPAD * HPAD) return;
  int n = idx / HPAD, k = idx - n * HPAD;
  float v = (n < HDIM && k < HDIM) ? W2[(size_t)k * HDIM + n] : 0.0f;
  unsigned short h = bf16_rne(v);
  hi[idx] = h;
  lo[idx] = bf16_rne(v - bf16_to_f32(h));
}

// ---------- fused pair-MLP: 1 wave = 64 pairs (4x16 sub-tiles) ----------
__global__ __launch_bounds__(128) void pair_mlp_kernel(
    const float* __restrict__ E,  const float* __restrict__ b1,
    const float* __restrict__ b2, const float* __restrict__ W3,
    const float* __restrict__ b3,
    const unsigned short* __restrict__ w1t_hi, const unsigned short* __restrict__ w1t_lo,
    const unsigned short* __restrict__ w2t_hi, const unsigned short* __restrict__ w2t_lo,
    float* __restrict__ out) {
  __shared__ float lds_h[4][16][168];   // per-wave 16x160 (stride 168) f32 bounce tile

  const int lane = threadIdx.x & 31;
  const int wave = threadIdx.x >> 5;
  const int tile = blockIdx.x * 4 + wave;          // [0, 2292)

  const int row  = lane & 15;                      // A-matrix M row for this lane
  const int hi16 = (lane >> 4) & 1;                // upper-half lanes own K+16..K+31
  const int c0   = hi16 * 16;
  const int koff = hi16 * 16;

  // Per sub-tile pair decode (16-pair sub-tiles never straddle batches).
  const float* ei[MSUB];
  const float* ej[MSUB];
  int bb[MSUB], ii[MSUB], jj[MSUB];
#pragma unroll
  for (int m = 0; m < MSUB; ++m) {
    int g = tile * 64 + m * 16 + row;
    int b = g / PAIRS_PER_B;
    int p = g - b * PAIRS_PER_B;                   // p = i*(i-1)/2 + j, j < i
    int i = (int)((1.0f + sqrtf(1.0f + 8.0f * (float)p)) * 0.5f);
    while (i * (i - 1) / 2 > p) --i;
    while ((i + 1) * i / 2 <= p) ++i;
    int j = p - i * (i - 1) / 2;
    bb[m] = b; ii[m] = i; jj[m] = j;
    ei[m] = E + ((size_t)b * NSEQ + i) * DDIM;
    ej[m] = E + ((size_t)b * NSEQ + j) * DDIM;
  }

  // Biases in registers (reused by all 4 sub-tiles).
  float bias1[NTILES], bias2[NTILES];
#pragma unroll
  for (int n = 0; n < NTILES; ++n) {
    int ncol = n * 16 + row;
    bias1[n] = (ncol < HDIM) ? b1[ncol] : 0.0f;
    bias2[n] = (ncol < HDIM) ? b2[ncol] : 0.0f;
  }

  const size_t colOff1 = (size_t)row * K1DIM;

  // ---- GEMM1: feat(64x2304) @ W1(2304x160); each B fragment feeds 12 WMMAs ----
  v8f acc[MSUB][NTILES] = {};
  v16bf ah[MSUB], al[MSUB];
  float f[16], g[16];

#pragma unroll 1
  for (int s = 0; s < 24; ++s) {                   // feat = e_i
    int o = s * 32 + c0;
#pragma unroll
    for (int m = 0; m < MSUB; ++m) {
      load8(ei[m] + o, f); load8(ei[m] + o + 8, f + 8);
      split_pack16(f, ah[m], al[m]);
    }
    const unsigned short* bh = w1t_hi + colOff1 + s * 32 + koff;
    const unsigned short* bl = w1t_lo + colOff1 + s * 32 + koff;
#pragma unroll
    for (int n = 0; n < NTILES; ++n) {
      v16bf Bh = __builtin_bit_cast(v16bf, *(const v8u*)(bh + (size_t)n * 16 * K1DIM));
      v16bf Bl = __builtin_bit_cast(v16bf, *(const v8u*)(bl + (size_t)n * 16 * K1DIM));
#pragma unroll
      for (int m = 0; m < MSUB; ++m)
        acc[m][n] = wmma3v(acc[m][n], ah[m], al[m], Bh, Bl);
    }
  }
#pragma unroll 1
  for (int s = 0; s < 24; ++s) {                   // feat = e_j
    int o = s * 32 + c0;
#pragma unroll
    for (int m = 0; m < MSUB; ++m) {
      load8(ej[m] + o, f); load8(ej[m] + o + 8, f + 8);
      split_pack16(f, ah[m], al[m]);
    }
    const unsigned short* bh = w1t_hi + colOff1 + (s + 24) * 32 + koff;
    const unsigned short* bl = w1t_lo + colOff1 + (s + 24) * 32 + koff;
#pragma unroll
    for (int n = 0; n < NTILES; ++n) {
      v16bf Bh = __builtin_bit_cast(v16bf, *(const v8u*)(bh + (size_t)n * 16 * K1DIM));
      v16bf Bl = __builtin_bit_cast(v16bf, *(const v8u*)(bl + (size_t)n * 16 * K1DIM));
#pragma unroll
      for (int m = 0; m < MSUB; ++m)
        acc[m][n] = wmma3v(acc[m][n], ah[m], al[m], Bh, Bl);
    }
  }
#pragma unroll 1
  for (int s = 0; s < 24; ++s) {                   // feat = e_i * e_j
    int o = s * 32 + c0;
#pragma unroll
    for (int m = 0; m < MSUB; ++m) {
      load8(ei[m] + o, f); load8(ei[m] + o + 8, f + 8);
      load8(ej[m] + o, g); load8(ej[m] + o + 8, g + 8);
#pragma unroll
      for (int t = 0; t < 16; ++t) f[t] *= g[t];
      split_pack16(f, ah[m], al[m]);
    }
    const unsigned short* bh = w1t_hi + colOff1 + (s + 48) * 32 + koff;
    const unsigned short* bl = w1t_lo + colOff1 + (s + 48) * 32 + koff;
#pragma unroll
    for (int n = 0; n < NTILES; ++n) {
      v16bf Bh = __builtin_bit_cast(v16bf, *(const v8u*)(bh + (size_t)n * 16 * K1DIM));
      v16bf Bl = __builtin_bit_cast(v16bf, *(const v8u*)(bl + (size_t)n * 16 * K1DIM));
#pragma unroll
      for (int m = 0; m < MSUB; ++m)
        acc[m][n] = wmma3v(acc[m][n], ah[m], al[m], Bh, Bl);
    }
  }

  // ---- per sub-tile: GEMM2 + layer3 through the per-wave LDS bounce tile ----
  float (*h)[168] = lds_h[wave];
  const int mrow = hi16 * 8;
  const size_t colOff2 = (size_t)row * HPAD;

#pragma unroll
  for (int m = 0; m < MSUB; ++m) {
    // h1 = relu(acc + b1) -> LDS (C layout: VGPR r = row r (+8 upper lanes), N = lane%16)
#pragma unroll
    for (int n = 0; n < NTILES; ++n) {
      int ncol = n * 16 + row;
#pragma unroll
      for (int r = 0; r < 8; ++r) {
        float v = acc[m][n][r] + bias1[n];
        h[mrow + r][ncol] = v > 0.0f ? v : 0.0f;
      }
    }

    // GEMM2: h1(16x160) @ W2(160x160), 5 K-slabs of 32
    v8f acc2[NTILES] = {};
#pragma unroll 1
    for (int s = 0; s < 5; ++s) {
      int o = s * 32 + c0;
      float4 a0 = *(const float4*)&h[row][o];
      float4 a1 = *(const float4*)&h[row][o + 4];
      float4 a2 = *(const float4*)&h[row][o + 8];
      float4 a3 = *(const float4*)&h[row][o + 12];
      f[0]  = a0.x; f[1]  = a0.y; f[2]  = a0.z; f[3]  = a0.w;
      f[4]  = a1.x; f[5]  = a1.y; f[6]  = a1.z; f[7]  = a1.w;
      f[8]  = a2.x; f[9]  = a2.y; f[10] = a2.z; f[11] = a2.w;
      f[12] = a3.x; f[13] = a3.y; f[14] = a3.z; f[15] = a3.w;
      v16bf ah2, al2;
      split_pack16(f, ah2, al2);
      const unsigned short* bh = w2t_hi + colOff2 + s * 32 + koff;
      const unsigned short* bl = w2t_lo + colOff2 + s * 32 + koff;
#pragma unroll
      for (int n = 0; n < NTILES; ++n) {
        v16bf Bh = __builtin_bit_cast(v16bf, *(const v8u*)(bh + (size_t)n * 16 * HPAD));
        v16bf Bl = __builtin_bit_cast(v16bf, *(const v8u*)(bl + (size_t)n * 16 * HPAD));
        acc2[n] = wmma3v(acc2[n], ah2, al2, Bh, Bl);
      }
    }

    // h2 = relu(acc2 + b2) -> LDS (same-wave DS ops are in-order)
#pragma unroll
    for (int n = 0; n < NTILES; ++n) {
      int ncol = n * 16 + row;
#pragma unroll
      for (int r = 0; r < 8; ++r) {
        float v = acc2[n][r] + bias2[n];
        h[mrow + r][ncol] = v > 0.0f ? v : 0.0f;
      }
    }

    // layer 3: scalar score per pair, scatter to logits
    if (lane < 16) {
      float s3 = b3[0];
      for (int k = 0; k < HDIM; ++k) s3 += h[row][k] * W3[k];
      out[((size_t)bb[m] * NSEQ + ii[m]) * NSEQ + jj[m]] = s3;
    }
  }
}

// ---------- masked row softmax (1 wave per row, in-place on out) ----------
__global__ __launch_bounds__(32) void softmax_kernel(float* __restrict__ out) {
  const int bid = blockIdx.x;
  const int b = bid / NSEQ;
  const int i = bid - b * NSEQ;
  float* rowp = out + ((size_t)b * NSEQ + i) * NSEQ;
  const int lane = threadIdx.x;

  float v[6];
  int cnt = 0;
  float m = (lane == 0) ? 0.0f : -3.4e38f;   // lane 0 owns the implicit 0 diagonal
  for (int j = lane; j < i; j += 32) {
    float x = rowp[j];
    v[cnt++] = x;
    m = fmaxf(m, x);
  }
#pragma unroll
  for (int off = 16; off >= 1; off >>= 1) m = fmaxf(m, __shfl_xor(m, off, 32));

  float sum = (lane == 0) ? expf(-m) : 0.0f;
  for (int c = 0; c < cnt; ++c) sum += expf(v[c] - m);
#pragma unroll
  for (int off = 16; off >= 1; off >>= 1) sum += __shfl_xor(sum, off, 32);
  float inv = 1.0f / sum;

  int c = 0;
  for (int j = lane; j < i; j += 32) rowp[j] = expf(v[c++] - m) * inv;
  if (lane == 0) rowp[i] = expf(-m) * inv;
  for (int j = i + 1 + lane; j < NSEQ; j += 32) rowp[j] = PAD_VAL;
}

extern "C" void kernel_launch(void* const* d_in, const int* in_sizes, int n_in,
                              void* d_out, int out_size, void* d_ws, size_t ws_size,
                              hipStream_t stream) {
  (void)in_sizes; (void)n_in; (void)out_size; (void)ws_size;
  const float* E  = (const float*)d_in[0];
  const float* W1 = (const float*)d_in[1];
  const float* b1 = (const float*)d_in[2];
  const float* W2 = (const float*)d_in[3];
  const float* b2 = (const float*)d_in[4];
  const float* W3 = (const float*)d_in[5];
  const float* b3 = (const float*)d_in[6];
  float* out = (float*)d_out;

  char* ws = (char*)d_ws;
  unsigned short* w1t_hi = (unsigned short*)(ws);                 // 160*2304*2 = 737280 B
  unsigned short* w1t_lo = (unsigned short*)(ws + 737280);        // 737280 B
  unsigned short* w2t_hi = (unsigned short*)(ws + 1474560);       // 160*160*2 = 51200 B
  unsigned short* w2t_lo = (unsigned short*)(ws + 1525760);       // 51200 B -> 1576960 total

  prep_w1<<<(HPAD * K1DIM + 255) / 256, 256, 0, stream>>>(W1, w1t_hi, w1t_lo);
  prep_w2<<<(HPAD * HPAD + 255) / 256, 256, 0, stream>>>(W2, w2t_hi, w2t_lo);
  pair_mlp_kernel<<<TOTAL_M64 / 4, 128, 0, stream>>>(
      E, b1, b2, W3, b3, w1t_hi, w1t_lo, w2t_hi, w2t_lo, out);
  softmax_kernel<<<BATCH * NSEQ, 32, 0, stream>>>(out);
}
